// RnnHSGModel_3186865734211
// MI455X (gfx1250) — compile-verified, matrix operand order checked by reading, not run on hardware
//
#include <hip/hip_runtime.h>
#include <hip/hip_bf16.h>
#include <math.h>

// ---------------- constants ----------------
#define BB 256           // batch (docs)
#define LL 256           // max timesteps
#define HH 256           // hidden
#define GG 768           // 3*H gates
#define BL (BB * LL)     // 65536 rows

typedef __attribute__((ext_vector_type(16))) __bf16 v16bf;
typedef __attribute__((ext_vector_type(8)))  float  v8f;

union Frag16 {
    uint4 u[2];
    v16bf v;
};

__device__ __forceinline__ v16bf load_frag(const unsigned short* p) {
    Frag16 f;
    f.u[0] = *(const uint4*)(p);
    f.u[1] = *(const uint4*)(p + 16);
    return f.v;
}

__device__ __forceinline__ unsigned short f2bf(float f) {
    __bf16 h = (__bf16)f;
    return __builtin_bit_cast(unsigned short, h);
}

__device__ __forceinline__ float sigmoidf_(float x) {
    return 1.0f / (1.0f + __expf(-x));
}

// CDNA5 async global->LDS copy (ASYNCcnt-tracked). LDS dest addr in a VGPR,
// 64-bit global addr in a VGPR pair. See CDNA5 ISA.
__device__ __forceinline__ void async_load_b128(unsigned lds_off, const void* gptr) {
    asm volatile("global_load_async_to_lds_b128 %0, %1, off"
                 :: "v"(lds_off), "v"((unsigned long long)(uintptr_t)gptr)
                 : "memory");
}
__device__ __forceinline__ void wait_asynccnt0() {
    asm volatile("s_wait_asynccnt 0x0" ::: "memory");
}

// ---------------- f32 -> bf16 conversion (vector x4) ----------------
__global__ void __launch_bounds__(256) cvt_f32_bf16(const float* __restrict__ s,
                                                    unsigned short* __restrict__ d,
                                                    int n) {
    int i = (blockIdx.x * 256 + threadIdx.x) * 4;
    if (i < n) {
        float4 v = *(const float4*)(s + i);
        ushort4 o;
        o.x = f2bf(v.x); o.y = f2bf(v.y); o.z = f2bf(v.z); o.w = f2bf(v.w);
        *(ushort4*)(d + i) = o;
    }
}

// ---------------- WMMA GEMM: C[dir][M,N] = A[M,K] @ W[dir][N,K]^T + bias[dir][N] ----------------
__global__ void __launch_bounds__(256) gemm_bf16_xg(
    const unsigned short* __restrict__ A,   // [M,K] bf16 row-major
    const unsigned short* __restrict__ W,   // [2][N,K] bf16 row-major
    const float* __restrict__ bias,         // [2][N]
    float* __restrict__ C,                  // [2][M,N]
    int M, int N, int K) {
    const int dir = blockIdx.y;
    W    += (size_t)dir * N * K;
    bias += (size_t)dir * N;
    C    += (size_t)dir * M * N;

    const int wave = (blockIdx.x * 256 + threadIdx.x) >> 5;
    const int lane = threadIdx.x & 31;
    const int nt64 = N >> 6;
    const int mt   = wave / nt64;
    const int nt   = wave % nt64;
    if (mt * 16 >= M) return;  // wave-uniform guard

    const int koff = (lane >> 4) << 3;                   // 0 or 8 (ISA bf16 A/B layout)
    const unsigned short* arow = A + (size_t)(mt * 16 + (lane & 15)) * K + koff;
    const unsigned short* brow = W + (size_t)(nt * 64 + (lane & 15)) * K + koff;
    const size_t wstride = (size_t)16 * K;

    v8f acc[4];
    acc[0] = v8f{}; acc[1] = v8f{}; acc[2] = v8f{}; acc[3] = v8f{};

    for (int k = 0; k < K; k += 32) {
        v16bf a = load_frag(arow + k);
        v16bf b[4];
#pragma unroll
        for (int sub = 0; sub < 4; ++sub)
            b[sub] = load_frag(brow + wstride * sub + k);
#pragma unroll
        for (int sub = 0; sub < 4; ++sub)
            acc[sub] = __builtin_amdgcn_wmma_f32_16x16x32_bf16(
                false, a, false, b[sub], (short)0, acc[sub], false, false);
    }

#pragma unroll
    for (int sub = 0; sub < 4; ++sub) {
        int nc = nt * 64 + sub * 16 + (lane & 15);
        float bv = bias[nc];
#pragma unroll
        for (int r = 0; r < 8; ++r) {
            int m = r + ((lane >> 4) << 3);
            C[(size_t)(mt * 16 + m) * N + nc] = acc[sub][r] + bv;
        }
    }
}

// ---------------- Bi-GRU recurrent scan ----------------
// grid = (B/16, 2 dirs), block = 256 (8 waves). Block owns 16 batch rows.
// Pipeline per step: [async prefetch of xg(s) overlaps] WMMA hg = h @ Whh^T,
// then s_wait_asynccnt + barrier, then gate math from LDS.
__global__ void __launch_bounds__(256) gru_scan(
    const unsigned short* __restrict__ Whh,  // [2][768][256] bf16
    const float* __restrict__ bhh,           // [2][768]
    const float* __restrict__ Xg,            // [2][B*L][768] (b_ih already added)
    const int* __restrict__ lengths,         // [B]
    float* __restrict__ Y) {                 // [B*L][512]
    const int dir   = blockIdx.y;
    const int btile = blockIdx.x;
    Whh += (size_t)dir * GG * HH;
    bhh += dir * GG;
    Xg  += (size_t)dir * BL * GG;

    __shared__ unsigned short hbf[16][264];  // bf16 h, padded vs bank conflicts
    __shared__ float          hg[16][772];   // gate pre-activations, padded
    __shared__ float          xgs[16][768];  // async-staged xg for current step
    __shared__ int            slen[16];

    const int tid  = threadIdx.x;
    const int wave = tid >> 5;
    const int lane = tid & 31;
    const int koff = (lane >> 4) << 3;

    for (int i = tid; i < 16 * 256; i += 256)
        hbf[i >> 8][i & 255] = f2bf(0.0f);
    if (tid < 16) slen[tid] = lengths[btile * 16 + tid];
    __syncthreads();

    const unsigned xg_lds = (unsigned)(uintptr_t)(&xgs[0][0]);

    // per-thread f32 hidden state: thread j owns column j of all 16 rows
    float hreg[16];
#pragma unroll
    for (int r = 0; r < 16; ++r) hreg[r] = 0.0f;

    const int j  = tid;              // gate column owned by this thread (0..255)
    const float br = bhh[j];
    const float bz = bhh[256 + j];
    const float bn = bhh[512 + j];

    // issue async prefetch of xg for step `sn` into the staging buffer
    auto prefetch_xg = [&](int sn) {
#pragma unroll
        for (int i = 0; i < 12; ++i) {
            int c   = tid + 256 * i;        // 0..3071 (16B chunks; 192 per row)
            int row = c / 192;
            int cb  = (c % 192) * 16;       // byte offset within row
            int lenb = slen[row];
            int o = (sn < lenb) ? (dir ? (lenb - 1 - sn) : sn) : sn;
            const char* g = (const char*)(Xg + ((size_t)(btile * 16 + row) * LL + o) * GG) + cb;
            async_load_b128(xg_lds + (unsigned)(row * 3072 + cb), g);
        }
    };

    prefetch_xg(0);   // prologue

    for (int s = 0; s < LL; ++s) {
        // ---- phase 1: hg = h @ Whh^T (WMMA); xg(s) async loads in flight ----
#pragma unroll
        for (int i = 0; i < 6; ++i) {
            const int nb = (wave * 6 + i) * 16;
            const unsigned short* brow = Whh + (size_t)(nb + (lane & 15)) * HH + koff;
            v8f acc = v8f{};
#pragma unroll
            for (int kt = 0; kt < 8; ++kt) {
                v16bf a = load_frag(&hbf[lane & 15][kt * 32 + koff]);
                v16bf b = load_frag(brow + kt * 32);
                acc = __builtin_amdgcn_wmma_f32_16x16x32_bf16(
                    false, a, false, b, (short)0, acc, false, false);
            }
#pragma unroll
            for (int r = 0; r < 8; ++r) {
                int m = r + ((lane >> 4) << 3);
                hg[m][nb + (lane & 15)] = acc[r];
            }
        }
        wait_asynccnt0();   // retire this wave's xg(s) async loads
        __syncthreads();    // hg + everyone's xg(s) now visible

        // ---- phase 2: gates + h update + y write (all operands in LDS/regs) ----
#pragma unroll
        for (int row = 0; row < 16; ++row) {
            const int lenb  = slen[row];
            const bool valid = (s < lenb);
            const int o = valid ? (dir ? (lenb - 1 - s) : s) : s;
            const size_t base = (size_t)(btile * 16 + row) * LL + o;
            float hprev = hreg[row];
            float hnew = hprev, yv = 0.0f;
            if (valid) {
                float rg = sigmoidf_(xgs[row][j]       + hg[row][j]       + br);
                float zg = sigmoidf_(xgs[row][256 + j] + hg[row][256 + j] + bz);
                float ng = tanhf(xgs[row][512 + j] + rg * (hg[row][512 + j] + bn));
                hnew = (1.0f - zg) * ng + zg * hprev;
                yv = hnew;
            }
            hreg[row] = hnew;
            hbf[row][j] = f2bf(hnew);
            Y[base * 512 + dir * 256 + j] = yv;
        }
        __syncthreads();    // xgs consumed; safe to overwrite

        // ---- issue prefetch for next step; overlaps next phase-1 WMMA ----
        prefetch_xg((s + 1) & (LL - 1));   // wraps to dummy load at s=L-1 (never read)
    }
}

// ---------------- MLP head: leaky_relu(y@w1^T+b1) -> elu(.@w2^T+b2), masked ----------------
__global__ void __launch_bounds__(256) head_kernel(
    const unsigned short* __restrict__ Ybf,  // [B*L][512] bf16
    const unsigned short* __restrict__ W1,   // [256][512] bf16
    const float* __restrict__ b1,            // [256]
    const float* __restrict__ w2,            // [2][256]
    const float* __restrict__ b2,            // [2]
    const int* __restrict__ lengths,         // [B]
    float* __restrict__ out) {               // [B*L][2]
    __shared__ float h1[16][256];
    __shared__ float red[16][2];

    const int mt   = blockIdx.x;
    const int wave = threadIdx.x >> 5;
    const int lane = threadIdx.x & 31;
    const int koff = (lane >> 4) << 3;

    const unsigned short* arow = Ybf + (size_t)(mt * 16 + (lane & 15)) * 512 + koff;

#pragma unroll
    for (int i = 0; i < 2; ++i) {
        const int nt = wave * 2 + i;
        const unsigned short* brow = W1 + (size_t)(nt * 16 + (lane & 15)) * 512 + koff;
        v8f acc = v8f{};
        for (int k = 0; k < 512; k += 32) {
            v16bf a = load_frag(arow + k);
            v16bf b = load_frag(brow + k);
            acc = __builtin_amdgcn_wmma_f32_16x16x32_bf16(
                false, a, false, b, (short)0, acc, false, false);
        }
#pragma unroll
        for (int r = 0; r < 8; ++r) {
            int m = r + ((lane >> 4) << 3);
            int n = nt * 16 + (lane & 15);
            float v = acc[r] + b1[n];
            h1[m][n] = (v > 0.0f) ? v : 0.01f * v;   // leaky relu
        }
    }
    if (threadIdx.x < 32) red[threadIdx.x >> 1][threadIdx.x & 1] = 0.0f;
    __syncthreads();

    const int r2 = threadIdx.x >> 4;   // row 0..15
    const int p  = threadIdx.x & 15;   // 16-way partial
    float s0 = 0.0f, s1 = 0.0f;
    for (int jj = p * 16; jj < p * 16 + 16; ++jj) {
        float hv = h1[r2][jj];
        s0 += hv * w2[jj];
        s1 += hv * w2[256 + jj];
    }
    atomicAdd(&red[r2][0], s0);
    atomicAdd(&red[r2][1], s1);
    __syncthreads();

    if (threadIdx.x < 32) {
        int rr = threadIdx.x >> 1;
        int o  = threadIdx.x & 1;
        int grow = mt * 16 + rr;
        int b = grow >> 8;      // / L
        int t = grow & 255;     // % L
        float v = red[rr][o] + b2[o];
        v = (v > 0.0f) ? v : (__expf(v) - 1.0f);     // elu, alpha=1
        if (t >= lengths[b]) v = 0.0f;
        out[(size_t)grow * 2 + o] = v;
    }
}

// ---------------- host launch ----------------
extern "C" void kernel_launch(void* const* d_in, const int* in_sizes, int n_in,
                              void* d_out, int out_size, void* d_ws, size_t ws_size,
                              hipStream_t stream) {
    const float* sent   = (const float*)d_in[0];
    const int*   lens   = (const int*)d_in[1];
    const float* w_ih0  = (const float*)d_in[2];
    const float* w_hh0  = (const float*)d_in[3];
    const float* b_ih0  = (const float*)d_in[4];
    const float* b_hh0  = (const float*)d_in[5];
    const float* w_ih   = (const float*)d_in[6];
    const float* w_hh   = (const float*)d_in[7];
    const float* b_ih   = (const float*)d_in[8];
    const float* b_hh   = (const float*)d_in[9];
    const float* w1     = (const float*)d_in[10];
    const float* b1     = (const float*)d_in[11];
    const float* w2     = (const float*)d_in[12];
    const float* b2     = (const float*)d_in[13];
    float* out = (float*)d_out;

    // workspace carve (256B aligned)
    char* p = (char*)d_ws;
    auto carve = [&](size_t bytes) -> void* {
        void* r = (void*)p;
        p += (bytes + 255) & ~(size_t)255;
        return r;
    };
    unsigned short* Xbf    = (unsigned short*)carve((size_t)BL * 512 * 2);
    unsigned short* Wih0bf = (unsigned short*)carve((size_t)2 * GG * 256 * 2);
    unsigned short* Whh0bf = (unsigned short*)carve((size_t)2 * GG * 256 * 2);
    unsigned short* Wihbf  = (unsigned short*)carve((size_t)3 * 2 * GG * 512 * 2);
    unsigned short* Whhbf  = (unsigned short*)carve((size_t)3 * 2 * GG * 256 * 2);
    unsigned short* W1bf   = (unsigned short*)carve((size_t)256 * 512 * 2);
    float*          Xg     = (float*)carve((size_t)2 * BL * GG * 4);
    float*          Yb     = (float*)carve((size_t)BL * 512 * 4);

    auto cvt = [&](const float* s, unsigned short* d, int n) {
        int blocks = (n / 4 + 255) / 256;
        cvt_f32_bf16<<<blocks, 256, 0, stream>>>(s, d, n);
    };

    // weight conversions (once per launch)
    cvt(w_ih0, Wih0bf, 2 * GG * 256);
    cvt(w_hh0, Whh0bf, 2 * GG * 256);
    cvt(w_ih,  Wihbf,  3 * 2 * GG * 512);
    cvt(w_hh,  Whhbf,  3 * 2 * GG * 256);
    cvt(w1,    W1bf,   256 * 512);
    // layer-0 input
    cvt(sent, Xbf, BL * 256);

    const int gemm_blocks = ((BL / 16) * (GG / 64)) / 8;  // 6144

    for (int layer = 0; layer < 4; ++layer) {
        const int K = (layer == 0) ? 256 : 512;
        const unsigned short* Wih_l = (layer == 0) ? Wih0bf : (Wihbf + (size_t)(layer - 1) * 2 * GG * 512);
        const unsigned short* Whh_l = (layer == 0) ? Whh0bf : (Whhbf + (size_t)(layer - 1) * 2 * GG * 256);
        const float* bih_l = (layer == 0) ? b_ih0 : (b_ih + (size_t)(layer - 1) * 2 * GG);
        const float* bhh_l = (layer == 0) ? b_hh0 : (b_hh + (size_t)(layer - 1) * 2 * GG);

        gemm_bf16_xg<<<dim3(gemm_blocks, 2), 256, 0, stream>>>(
            Xbf, Wih_l, bih_l, Xg, BL, GG, K);
        gru_scan<<<dim3(BB / 16, 2), 256, 0, stream>>>(
            Whh_l, bhh_l, Xg, lens, Yb);
        cvt(Yb, Xbf, BL * 512);  // next-layer input (or head input after layer 3)
    }

    head_kernel<<<BL / 16, 256, 0, stream>>>(Xbf, W1bf, b1, w2, b2, lens, out);
}
